// SeqRecModel_24060406792478
// MI455X (gfx1250) — matrix operand order repaired and checked by reading, not compiled, not run on hardware
//
#include <hip/hip_runtime.h>
#include <hip/hip_bf16.h>

#define B_   256
#define S_   16
#define I_   64
#define D_   128
#define T_   65                  // I+1 (BOS prepended)
#define M_   (B_*T_)             // 16640 GEMM rows
#define ROWS_ 64                 // rows per block (4 m-tiles per wave)
#define MB_  (M_/ROWS_)          // 260 blocks, exact
#define BOS_ 100000

typedef __attribute__((ext_vector_type(16))) _Float16 v16h;
typedef __attribute__((ext_vector_type(8)))  _Float16 v8h;
typedef __attribute__((ext_vector_type(4)))  _Float16 v4h;
typedef __attribute__((ext_vector_type(8)))  float    v8f;

// -------------------------------------------------------------------------
// once-per-launch init: h0 = 0, u_type -> out tail (as float)
__global__ void init_kernel(float* __restrict__ h, const int* __restrict__ u_type,
                            float* __restrict__ out_utype) {
    int i = blockIdx.x * blockDim.x + threadIdx.x;
    if (i < B_ * D_) h[i] = 0.f;
    if (i < B_)      out_utype[i] = (float)u_type[i];
}

// -------------------------------------------------------------------------
// Pack conv weights (OIH, tap reversed for causal shift) and fus_w[:, :128]
// into the WMMA B-matrix lane layout:
//   half at [group*512 + lane*16 + e] = W[n = nt*16 + (lane&15),
//                                        k = c*32 + ((lane>>4)<<3) + (e&7) + ((e>>3)<<4)]
// group = ((tap*4 + c)*8 + nt) for conv, (c*8 + nt) for fusion.
__global__ void pack_weights_kernel(const float* __restrict__ w0,
                                    const float* __restrict__ w1,
                                    const float* __restrict__ w2,
                                    const float* __restrict__ fw,
                                    _Float16* __restrict__ convWp,
                                    _Float16* __restrict__ fusWp) {
    int i = blockIdx.x * blockDim.x + threadIdx.x;
    const int perLayer = 3 * 4 * 8 * 512;              // 49152 halves
    if (i < 3 * perLayer) {
        int layer = i / perLayer, r = i % perLayer;
        int group = r >> 9, li = r & 511;
        int l = li >> 4, e = li & 15;
        int nt = group & 7, c = (group >> 3) & 3, kt = group >> 5;   // shift = kt*dil
        int o  = nt * 16 + (l & 15);
        int ch = c * 32 + ((l >> 4) << 3) + (e & 7) + ((e >> 3) << 4);
        const float* w = (layer == 0) ? w0 : (layer == 1) ? w1 : w2;
        convWp[i] = (_Float16)w[o * (D_ * 3) + ch * 3 + (2 - kt)];   // tap reversal
    }
    if (i < 4 * 8 * 512) {
        int group = i >> 9, li = i & 511;
        int l = li >> 4, e = li & 15;
        int nt = group & 7, c = group >> 3;
        int o  = nt * 16 + (l & 15);
        int ch = c * 32 + ((l >> 4) << 3) + (e & 7) + ((e >> 3) << 4);
        fusWp[i] = (_Float16)fw[o * (2 * D_) + ch];                  // first-half cols
    }
}

// -------------------------------------------------------------------------
// per-step: lengths[b] + batch max_len (single block, B_=256 threads)
__global__ void lengths_kernel(const int* __restrict__ item_id, int s,
                               int* __restrict__ lengths, int* __restrict__ maxlen) {
    __shared__ int red[B_];
    int b = threadIdx.x;
    const int* row = item_id + (b * S_ + s) * I_;
    int cnt = 0;
    for (int i = 0; i < I_; i++) { int v = row[i]; if (v < 0) v = 0; cnt += (v != 0); }
    lengths[b] = cnt;
    red[b] = cnt;
    __syncthreads();
    for (int off = B_ / 2; off > 0; off >>= 1) {
        if (b < off) red[b] = max(red[b], red[b + off]);
        __syncthreads();
    }
    if (b == 0) *maxlen = red[0];
}

// -------------------------------------------------------------------------
// embedding gather + loss-mask write + masked mean (block = one batch row)
__global__ void gather_kernel(const int* __restrict__ item_id, int s,
                              const int* __restrict__ lengths,
                              const int* __restrict__ maxlen,
                              const int* __restrict__ eval_from,
                              const float* __restrict__ emb,
                              float* __restrict__ e, float* __restrict__ mean_emb,
                              float* __restrict__ out_masks) {
    int b = blockIdx.x, d = threadIdx.x;
    int ml = *maxlen, len = lengths[b];
    bool allow = (s >= eval_from[b]);
    float sum = 0.f, cnt = 0.f;
    for (int t = 0; t < T_; t++) {
        int id;
        if (t == 0) id = BOS_;
        else { id = item_id[(b * S_ + s) * I_ + (t - 1)]; if (id < 0) id = 0; }
        if (t > ml) id = 0;                                   // batch-level truncation
        bool m = (id != 0) && !((t == len) && (len > 0));     // drop last-valid pos
        float v = emb[(long)id * D_ + d];                     // row 0 is all zeros
        e[((long)b * T_ + t) * D_ + d] = v;
        if (m) { sum += v; cnt += 1.f; }
        if (d == 0) out_masks[(b * S_ + s) * T_ + t] = (m && allow) ? 1.f : 0.f;
    }
    mean_emb[b * D_ + d] = sum / fmaxf(cnt, 1.f);
}

// -------------------------------------------------------------------------
// per-b fusion bias table: hW2b[b,n] = fus_b[n] + h[b,:] . fus_w[n, 128:256]
__global__ void hw2_kernel(const float* __restrict__ h, const float* __restrict__ fw,
                           const float* __restrict__ fb, float* __restrict__ hW2b) {
    __shared__ float hs[D_];
    int b = blockIdx.x, n = threadIdx.x;
    hs[n] = h[b * D_ + n];
    __syncthreads();
    float acc = fb[n];
    const float* wr = fw + n * (2 * D_) + D_;
    for (int j = 0; j < D_; j++) acc += hs[j] * wr[j];
    hW2b[b * D_ + n] = acc;
}

// -------------------------------------------------------------------------
// One TCN layer: Y = relu( sum_{kt} W[kt] @ X[t - kt*dil] + bias + X )
// Block = 64-row M-slab; 8 waves each own one 16-col N-tile and carry 4
// accumulators (4 m-tiles), so each B fragment is loaded once and reused
// 4x, and the 4 independent WMMA chains hide the D->C RAW latency.
__global__ void __launch_bounds__(256)
tcn_kernel(const float* __restrict__ X, float* __restrict__ Y,
           const _Float16* __restrict__ Wp, const float* __restrict__ bias, int dil) {
    __shared__ _Float16 lx[(ROWS_ + 8) * D_];     // rows R0-8 .. R0+63 (18 KB)
    int R0 = blockIdx.x * ROWS_;
    for (int q = threadIdx.x; q < (ROWS_ + 8) * (D_ / 4); q += 256) {
        int rowl = q >> 5;                        // 32 four-chan chunks per row
        int ch4  = (q & 31) << 2;
        int g = R0 - 8 + rowl;
        float4 v = make_float4(0.f, 0.f, 0.f, 0.f);
        if (g >= 0) v = *(const float4*)(X + (size_t)g * D_ + ch4);
        v4h h4;
        h4[0] = (_Float16)v.x; h4[1] = (_Float16)v.y;
        h4[2] = (_Float16)v.z; h4[3] = (_Float16)v.w;
        *(v4h*)(lx + rowl * D_ + ch4) = h4;
    }
    __syncthreads();

    int l = threadIdx.x & 31, w = threadIdx.x >> 5;
    int m16 = l & 15, hl = l >> 4, nb = w * 16;
    int tj[4];
    #pragma unroll
    for (int j = 0; j < 4; j++) tj[j] = (R0 + j * 16 + m16) % T_;

    v8f acc[4];                                   // C/D: M = r + hl*8, N = nb + m16
    #pragma unroll
    for (int j = 0; j < 4; j++)
        #pragma unroll
        for (int r = 0; r < 8; r++) {
            int mc = R0 + j * 16 + r + (hl << 3), nc = nb + m16;
            acc[j][r] = bias[nc] + X[(long)mc * D_ + nc];   // bias + residual
        }

    const v16h zero16 = {};
    #pragma unroll
    for (int kt = 0; kt < 3; kt++) {
        int shift = kt * dil;
        #pragma unroll
        for (int c = 0; c < 4; c++) {
            v16h bm = *(const v16h*)(Wp + ((((kt * 4 + c) * 8 + w)) << 9) + (l << 4));
            #pragma unroll
            for (int j = 0; j < 4; j++) {
                int rowl = j * 16 + m16 + 8 - shift;
                const _Float16* p = &lx[rowl * D_ + c * 32 + (hl << 3)];
                v8h lo = *(const v8h*)p;          // unconditional 16B ds loads
                v8h hi = *(const v8h*)(p + 16);
                v16h av;
                #pragma unroll
                for (int e2 = 0; e2 < 8; e2++) { av[e2] = lo[e2]; av[e2 + 8] = hi[e2]; }
                v16h a = (tj[j] >= shift) ? av : zero16;    // causal zero-pad
                acc[j] = __builtin_amdgcn_wmma_f32_16x16x32_f16(false, a, false, bm,
                                                      (short)0, acc[j], false, false);
            }
        }
    }

    #pragma unroll
    for (int j = 0; j < 4; j++)
        #pragma unroll
        for (int r = 0; r < 8; r++) {
            int mc = R0 + j * 16 + r + (hl << 3), nc = nb + m16;
            Y[(long)mc * D_ + nc] = fmaxf(acc[j][r], 0.f);
        }
}

// -------------------------------------------------------------------------
// Fusion: reps = (tc @ W1.T + hW2b[b,:]) * (t <= max_len), written to d_out
__global__ void __launch_bounds__(256)
fusion_kernel(const float* __restrict__ X, const float* __restrict__ hW2b,
              const int* __restrict__ maxlen, float* __restrict__ out, int s,
              const _Float16* __restrict__ Wp) {
    __shared__ _Float16 lx[ROWS_ * D_];           // 16 KB
    int R0 = blockIdx.x * ROWS_;
    for (int q = threadIdx.x; q < ROWS_ * (D_ / 4); q += 256) {
        int ch4 = q << 2;
        float4 v = *(const float4*)(X + (size_t)R0 * D_ + ch4);
        v4h h4;
        h4[0] = (_Float16)v.x; h4[1] = (_Float16)v.y;
        h4[2] = (_Float16)v.z; h4[3] = (_Float16)v.w;
        *(v4h*)(lx + ch4) = h4;
    }
    __syncthreads();

    int ml = *maxlen;
    int l = threadIdx.x & 31, w = threadIdx.x >> 5;
    int m16 = l & 15, hl = l >> 4, nb = w * 16;

    v8f acc[4];
    #pragma unroll
    for (int j = 0; j < 4; j++)
        #pragma unroll
        for (int r = 0; r < 8; r++) {
            int mc = R0 + j * 16 + r + (hl << 3);
            acc[j][r] = hW2b[(mc / T_) * D_ + nb + m16];
        }
    #pragma unroll
    for (int c = 0; c < 4; c++) {
        v16h bm = *(const v16h*)(Wp + ((c * 8 + w) << 9) + (l << 4));
        #pragma unroll
        for (int j = 0; j < 4; j++) {
            const _Float16* p = &lx[(j * 16 + m16) * D_ + c * 32 + (hl << 3)];
            v8h lo = *(const v8h*)p;
            v8h hi = *(const v8h*)(p + 16);
            v16h a;
            #pragma unroll
            for (int e2 = 0; e2 < 8; e2++) { a[e2] = lo[e2]; a[e2 + 8] = hi[e2]; }
            acc[j] = __builtin_amdgcn_wmma_f32_16x16x32_f16(false, a, false, bm,
                                                  (short)0, acc[j], false, false);
        }
    }
    #pragma unroll
    for (int j = 0; j < 4; j++)
        #pragma unroll
        for (int r = 0; r < 8; r++) {
            int mc = R0 + j * 16 + r + (hl << 3);
            int b = mc / T_, t = mc % T_;
            float v = (t <= ml) ? acc[j][r] : 0.f;
            out[(((long)b * S_ + s) * T_ + t) * D_ + nb + m16] = v;
        }
}

// -------------------------------------------------------------------------
// GRU step (tiny: plain f32 VALU). block = batch row, thread = output dim.
__global__ void gru_kernel(const float* __restrict__ x, const float* __restrict__ h,
                           const float* __restrict__ wih, const float* __restrict__ whh,
                           const float* __restrict__ bih, const float* __restrict__ bhh,
                           float* __restrict__ hout) {
    __shared__ float xs[D_], hs[D_];
    int b = blockIdx.x, d = threadIdx.x;
    xs[d] = x[b * D_ + d];
    hs[d] = h[b * D_ + d];
    __syncthreads();
    float gi[3], gh[3];
    #pragma unroll
    for (int g = 0; g < 3; g++) {
        float a1 = bih[g * D_ + d], a2 = bhh[g * D_ + d];
        const float* w1 = wih + (g * D_ + d) * D_;
        const float* w2 = whh + (g * D_ + d) * D_;
        for (int j = 0; j < D_; j++) { a1 += xs[j] * w1[j]; a2 += hs[j] * w2[j]; }
        gi[g] = a1; gh[g] = a2;
    }
    float r = 1.f / (1.f + expf(-(gi[0] + gh[0])));
    float z = 1.f / (1.f + expf(-(gi[1] + gh[1])));
    float n = tanhf(gi[2] + r * gh[2]);
    hout[b * D_ + d] = (1.f - z) * n + z * hs[d];
}

// -------------------------------------------------------------------------
extern "C" void kernel_launch(void* const* d_in, const int* in_sizes, int n_in,
                              void* d_out, int out_size, void* d_ws, size_t ws_size,
                              hipStream_t stream) {
    const int*   item_id   = (const int*)d_in[0];
    const int*   eval_from = (const int*)d_in[1];
    const int*   u_type    = (const int*)d_in[2];
    const float* emb       = (const float*)d_in[3];
    const float* cw0 = (const float*)d_in[4];  const float* cb0 = (const float*)d_in[5];
    const float* cw1 = (const float*)d_in[6];  const float* cb1 = (const float*)d_in[7];
    const float* cw2 = (const float*)d_in[8];  const float* cb2 = (const float*)d_in[9];
    const float* wih = (const float*)d_in[10]; const float* whh = (const float*)d_in[11];
    const float* bih = (const float*)d_in[12]; const float* bhh = (const float*)d_in[13];
    const float* fw  = (const float*)d_in[14]; const float* fb  = (const float*)d_in[15];

    float* ws = (float*)d_ws;
    size_t off = 0;
    float* e    = ws + off; off += (size_t)M_ * D_;
    float* bufA = ws + off; off += (size_t)M_ * D_;
    float* bufB = ws + off; off += (size_t)M_ * D_;
    float* mean = ws + off; off += B_ * D_;
    float* hA   = ws + off; off += B_ * D_;
    float* hB   = ws + off; off += B_ * D_;
    float* hW2b = ws + off; off += B_ * D_;
    _Float16* convWp = (_Float16*)(ws + off); off += (3 * 49152) / 2;   // f16 halves
    _Float16* fusWp  = (_Float16*)(ws + off); off += 16384 / 2;
    int* lengths = (int*)(ws + off); off += 256;
    int* maxlen  = (int*)(ws + off); off += 64;
    (void)ws_size; (void)in_sizes; (void)n_in; (void)out_size;

    float* out       = (float*)d_out;
    float* out_masks = out + (size_t)B_ * S_ * T_ * D_;
    float* out_utype = out_masks + (size_t)B_ * S_ * T_;

    init_kernel<<<(B_ * D_ + 255) / 256, 256, 0, stream>>>(hA, u_type, out_utype);
    pack_weights_kernel<<<(3 * 49152 + 255) / 256, 256, 0, stream>>>(
        cw0, cw1, cw2, fw, convWp, fusWp);

    const float* bias[3] = {cb0, cb1, cb2};
    float* hcur = hA; float* hnext = hB;
    for (int s = 0; s < S_; s++) {
        lengths_kernel<<<1, B_, 0, stream>>>(item_id, s, lengths, maxlen);
        gather_kernel<<<B_, D_, 0, stream>>>(item_id, s, lengths, maxlen, eval_from,
                                             emb, e, mean, out_masks);
        hw2_kernel<<<B_, D_, 0, stream>>>(hcur, fw, fb, hW2b);
        tcn_kernel<<<MB_, 256, 0, stream>>>(e,    bufA, convWp,             bias[0], 1);
        tcn_kernel<<<MB_, 256, 0, stream>>>(bufA, bufB, convWp + 49152,     bias[1], 2);
        tcn_kernel<<<MB_, 256, 0, stream>>>(bufB, bufA, convWp + 2 * 49152, bias[2], 4);
        fusion_kernel<<<MB_, 256, 0, stream>>>(bufA, hW2b, maxlen, out, s, fusWp);
        gru_kernel<<<B_, D_, 0, stream>>>(mean, hcur, wih, whh, bih, bhh, hnext);
        float* t = hcur; hcur = hnext; hnext = t;
    }
}
